// Eye_Center_38826504356073
// MI455X (gfx1250) — matrix operand order repaired and checked by reading, not compiled, not run on hardware
//
#include <hip/hip_runtime.h>
#include <hip/hip_bf16.h>
#include <math.h>

// ---------------------------------------------------------------------------
// MI455X (gfx1250, wave32) implementation.
// All GEMM-shaped math goes through v_wmma_f32_16x16x32_f16 (f32 accumulate).
// One wave (32 threads) per WMMA workgroup -> EXEC is all ones around WMMA.
// ---------------------------------------------------------------------------

typedef __attribute__((ext_vector_type(16))) _Float16 v16h;
typedef __attribute__((ext_vector_type(8)))  float    v8f;

#define ATT_SCALE 0.17677669529663687f   // (384/12)^-0.5 = 1/sqrt(32)
#define CDIM   384
#define QKVDIM 1152
#define NHEAD  12

__device__ __forceinline__ v8f wmma16(v16h a, v16h b, v8f c) {
  // D = A(16x32 f16) * B(32x16 f16) + C(16x16 f32)
  return __builtin_amdgcn_wmma_f32_16x16x32_f16(false, a, false, b, (short)0, c,
                                                false, false);
}

// A-fragment (16x32 f16): lane holds row = lane&15; half j -> K index:
__device__ __forceinline__ int a_col(int lane, int j) {
  return (j & 7) + ((j >> 3) << 4) + ((lane >> 4) << 3);
}
// B-fragment (32x16 f16): lane holds col = lane&15; half j -> K index:
__device__ __forceinline__ int b_k(int lane, int j) {
  return (j & 7) + ((j >> 3) << 4) + ((lane >> 4) << 3);
}
// C/D fragment (16x16 f32): c[r] at lane L is element (r + 8*(L>>4), L&15).

// 16-lane-group reductions (rows of a WMMA C fragment live in a 16-lane half).
__device__ __forceinline__ float grp_max16(float v) {
  v = fmaxf(v, __shfl_xor(v, 1, 32));
  v = fmaxf(v, __shfl_xor(v, 2, 32));
  v = fmaxf(v, __shfl_xor(v, 4, 32));
  v = fmaxf(v, __shfl_xor(v, 8, 32));
  return v;
}
__device__ __forceinline__ float grp_sum16(float v) {
  v += __shfl_xor(v, 1, 32);
  v += __shfl_xor(v, 2, 32);
  v += __shfl_xor(v, 4, 32);
  v += __shfl_xor(v, 8, 32);
  return v;
}

// ---------------------------------------------------------------------------
// 1) avgpool2 of x (B,64,64,C channel-last) -> xp16 (B*1024, 384) f16
// ---------------------------------------------------------------------------
__global__ void pool_kernel(const float* __restrict__ x, _Float16* __restrict__ xp) {
  int idx = blockIdx.x * 256 + threadIdx.x;       // < 2*1024*384
  int c = idx % CDIM;
  int t = idx / CDIM;
  int n = t % 1024;
  int b = t / 1024;
  int h = n >> 5, w = n & 31;
  size_t base = (((size_t)b * 64 + 2 * h) * 64 + 2 * w) * CDIM + c;
  float v = 0.25f * (x[base] + x[base + CDIM] +
                     x[base + 64 * CDIM] + x[base + 64 * CDIM + CDIM]);
  xp[idx] = (_Float16)v;
}

// D crop: x[:, 16:48, 16:48, :] -> (B*1024, 384) f16
__global__ void gather_d_kernel(const float* __restrict__ x, _Float16* __restrict__ xd) {
  int idx = blockIdx.x * 256 + threadIdx.x;       // < 2*1024*384
  int c = idx % CDIM;
  int t = idx / CDIM;
  int n = t % 1024;
  int b = t / 1024;
  int h = n >> 5, w = n & 31;
  xd[idx] = (_Float16)x[(((size_t)b * 64 + 16 + h) * 64 + 16 + w) * CDIM + c];
}

// crops of pooled 32x32 image -> (B*192, 384) f16
__global__ void gather_crop_kernel(const _Float16* __restrict__ xp,
                                   _Float16* __restrict__ out,
                                   int h0, int w0, int wd) {
  int idx = blockIdx.x * 256 + threadIdx.x;       // < 2*192*384
  int c = idx % CDIM;
  int t = idx / CDIM;
  int n = t % 192;
  int b = t / 192;
  int src = (h0 + n / wd) * 32 + (w0 + n % wd);
  out[idx] = xp[((size_t)b * 1024 + src) * CDIM + c];
}

// ---------------------------------------------------------------------------
// 2) Generic WMMA GEMM: Y(RxM) = X(RxK f16) @ W(KxM f32) + bias
// grid = (R/16, M/16), block = 32 (one wave)
// ---------------------------------------------------------------------------
template <bool HALF_OUT>
__global__ void gemm_bias_kernel(const _Float16* __restrict__ X,
                                 const float* __restrict__ W,
                                 const float* __restrict__ bias,
                                 void* __restrict__ Yv,
                                 int R, int K, int M) {
  int lane = threadIdx.x;
  int rb = blockIdx.x * 16;
  int nb = blockIdx.y * 16;
  int arow = rb + (lane & 15);
  int bcol = nb + (lane & 15);
  v8f acc = {};
  for (int kb = 0; kb < K; kb += 32) {
    v16h a, bf;
#pragma unroll
    for (int j = 0; j < 16; ++j) {
      a[j]  = X[(size_t)arow * K + kb + a_col(lane, j)];
      bf[j] = (_Float16)W[(size_t)(kb + b_k(lane, j)) * M + bcol];
    }
    acc = wmma16(a, bf, acc);
  }
  float bv = bias[bcol];
#pragma unroll
  for (int r = 0; r < 8; ++r) {
    int m = r + ((lane >> 4) << 3);
    float v = acc[r] + bv;
    if (HALF_OUT)
      ((_Float16*)Yv)[(size_t)(rb + m) * M + bcol] = (_Float16)v;
    else
      ((float*)Yv)[(size_t)(rb + m) * M + bcol] = v;
  }
}

// ---------------------------------------------------------------------------
// 3) Raw scaled scores S = scale * q @ k^T for the gw branch, written to d_out
// grid = (B*12, N/16, N/16), block = 32
// ---------------------------------------------------------------------------
__global__ void qk_scores_kernel(const _Float16* __restrict__ qkv,
                                 float* __restrict__ S, int N) {
  int lane = threadIdx.x;
  int bh = blockIdx.x;
  int b = bh / NHEAD, hd = bh % NHEAD;
  int rt = blockIdx.y * 16;
  int ct = blockIdx.z * 16;
  const _Float16* base = qkv + (size_t)b * N * QKVDIM;
  int qrow = rt + (lane & 15);
  int kcol = ct + (lane & 15);
  v16h a, kf;
#pragma unroll
  for (int j = 0; j < 16; ++j) {
    a[j]  = base[(size_t)qrow * QKVDIM + hd * 32 + a_col(lane, j)];
    kf[j] = base[(size_t)kcol * QKVDIM + CDIM + hd * 32 + b_k(lane, j)];
  }
  v8f c = {};
  c = wmma16(a, kf, c);
#pragma unroll
  for (int r = 0; r < 8; ++r) {
    int m = r + ((lane >> 4) << 3);
    S[((size_t)bh * N + rt + m) * N + kcol] = c[r] * ATT_SCALE;
  }
}

// In-place row softmax over N columns. grid = #rows, block = 256.
__global__ void softmax_rows_kernel(float* __restrict__ S, int N) {
  int tid = threadIdx.x;
  float* p = S + (size_t)blockIdx.x * N;
  __shared__ float red[256];
  float mx = -1e30f;
  for (int i = tid; i < N; i += 256) mx = fmaxf(mx, p[i]);
  red[tid] = mx; __syncthreads();
  for (int o = 128; o > 0; o >>= 1) {
    if (tid < o) red[tid] = fmaxf(red[tid], red[tid + o]);
    __syncthreads();
  }
  mx = red[0]; __syncthreads();
  float sm = 0.f;
  for (int i = tid; i < N; i += 256) sm += expf(p[i] - mx);
  red[tid] = sm; __syncthreads();
  for (int o = 128; o > 0; o >>= 1) {
    if (tid < o) red[tid] += red[tid + o];
    __syncthreads();
  }
  float inv = 1.f / red[0];
  for (int i = tid; i < N; i += 256) p[i] = expf(p[i] - mx) * inv;
}

// ---------------------------------------------------------------------------
// 4) Streaming attention (no NxN materialization) for the atten_shift branches.
// Pass 1: per-query-row running max & exp-sum (WMMA + shuffle reductions).
// Pass 2: O = softmax(S) @ V via WMMA; P transposed to A-layout through LDS.
// grid = (B*12, N/16), block = 32.
// ---------------------------------------------------------------------------
__global__ void attn_pass1_kernel(const _Float16* __restrict__ qkv,
                                  float* __restrict__ stats, int N) {
  int lane = threadIdx.x;
  int bh = blockIdx.x;
  int b = bh / NHEAD, hd = bh % NHEAD;
  int rbase = blockIdx.y * 16;
  const _Float16* base = qkv + (size_t)b * N * QKVDIM;
  int qrow = rbase + (lane & 15);
  v16h a;
#pragma unroll
  for (int j = 0; j < 16; ++j)
    a[j] = base[(size_t)qrow * QKVDIM + hd * 32 + a_col(lane, j)];
  float mr[8], lr[8];
#pragma unroll
  for (int r = 0; r < 8; ++r) { mr[r] = -1e30f; lr[r] = 0.f; }
  for (int ct = 0; ct < N; ct += 16) {
    v16h kf;
    int kcol = ct + (lane & 15);
#pragma unroll
    for (int j = 0; j < 16; ++j)
      kf[j] = base[(size_t)kcol * QKVDIM + CDIM + hd * 32 + b_k(lane, j)];
    v8f s = {};
    s = wmma16(a, kf, s);
#pragma unroll
    for (int r = 0; r < 8; ++r) {
      float val = s[r] * ATT_SCALE;
      float nm = fmaxf(mr[r], grp_max16(val));
      float gs = grp_sum16(expf(val - nm));
      lr[r] = lr[r] * expf(mr[r] - nm) + gs;
      mr[r] = nm;
    }
  }
  if ((lane & 15) == 0) {
#pragma unroll
    for (int r = 0; r < 8; ++r) {
      int row = rbase + r + ((lane >> 4) << 3);
      stats[((size_t)bh * N + row) * 2 + 0] = mr[r];
      stats[((size_t)bh * N + row) * 2 + 1] = lr[r];
    }
  }
}

__global__ void attn_pass2_kernel(const _Float16* __restrict__ qkv,
                                  const float* __restrict__ stats,
                                  _Float16* __restrict__ out, int N) {
  __shared__ _Float16 pl[16][32];
  int lane = threadIdx.x;
  int bh = blockIdx.x;
  int b = bh / NHEAD, hd = bh % NHEAD;
  int rbase = blockIdx.y * 16;
  const _Float16* base = qkv + (size_t)b * N * QKVDIM;
  int qrow = rbase + (lane & 15);
  v16h a;
#pragma unroll
  for (int j = 0; j < 16; ++j)
    a[j] = base[(size_t)qrow * QKVDIM + hd * 32 + a_col(lane, j)];
  float mr[8], inv[8];
#pragma unroll
  for (int r = 0; r < 8; ++r) {
    int row = rbase + r + ((lane >> 4) << 3);
    mr[r]  = stats[((size_t)bh * N + row) * 2 + 0];
    inv[r] = 1.f / stats[((size_t)bh * N + row) * 2 + 1];
  }
  v8f o0 = {}, o1 = {};
  for (int kb = 0; kb < N; kb += 32) {
    v8f s0 = {}, s1 = {};
    {
      v16h kf;
      int kcol = kb + (lane & 15);
#pragma unroll
      for (int j = 0; j < 16; ++j)
        kf[j] = base[(size_t)kcol * QKVDIM + CDIM + hd * 32 + b_k(lane, j)];
      s0 = wmma16(a, kf, s0);
      kcol += 16;
#pragma unroll
      for (int j = 0; j < 16; ++j)
        kf[j] = base[(size_t)kcol * QKVDIM + CDIM + hd * 32 + b_k(lane, j)];
      s1 = wmma16(a, kf, s1);
    }
    __syncthreads();   // WAR: previous iteration's reads of pl are done
#pragma unroll
    for (int r = 0; r < 8; ++r) {
      int row = r + ((lane >> 4) << 3);
      pl[row][(lane & 15)]      = (_Float16)expf(s0[r] * ATT_SCALE - mr[r]);
      pl[row][16 + (lane & 15)] = (_Float16)expf(s1[r] * ATT_SCALE - mr[r]);
    }
    __syncthreads();
    v16h pa, v0, v1;
#pragma unroll
    for (int j = 0; j < 16; ++j)
      pa[j] = pl[lane & 15][a_col(lane, j)];
#pragma unroll
    for (int j = 0; j < 16; ++j) {
      int key = kb + b_k(lane, j);
      v0[j] = base[(size_t)key * QKVDIM + 2 * CDIM + hd * 32 + (lane & 15)];
      v1[j] = base[(size_t)key * QKVDIM + 2 * CDIM + hd * 32 + 16 + (lane & 15)];
    }
    o0 = wmma16(pa, v0, o0);
    o1 = wmma16(pa, v1, o1);
  }
#pragma unroll
  for (int r = 0; r < 8; ++r) {
    int m = r + ((lane >> 4) << 3);
    size_t rowG = (size_t)b * N + rbase + m;
    out[rowG * CDIM + hd * 32 + (lane & 15)]      = (_Float16)(o0[r] * inv[r]);
    out[rowG * CDIM + hd * 32 + 16 + (lane & 15)] = (_Float16)(o1[r] * inv[r]);
  }
}

// ---------------------------------------------------------------------------
// 5) Epilogue: (B,N,C) -> (B,C,hdim,wdim), optional 2x align-corners upsample,
// zscore over (C*H) per (b,w), scatter to attn_all (B,64,64,C) in d_out.
// grid = (B, Wout), block = 256.
// ---------------------------------------------------------------------------
__device__ __forceinline__ float ep_sample(const float* __restrict__ sb,
                                           int hdim, int wdim, int up,
                                           int hp, int x0, int x1, float fx,
                                           int c) {
  int y0, y1; float fy;
  if (up) {
    float sy = hp * (float)(hdim - 1) / (float)(2 * hdim - 1);
    y0 = (int)floorf(sy);
    y1 = min(y0 + 1, hdim - 1);
    fy = sy - (float)y0;
  } else { y0 = y1 = hp; fy = 0.f; }
  float v00 = sb[(size_t)(y0 * wdim + x0) * CDIM + c];
  float v01 = sb[(size_t)(y0 * wdim + x1) * CDIM + c];
  float v10 = sb[(size_t)(y1 * wdim + x0) * CDIM + c];
  float v11 = sb[(size_t)(y1 * wdim + x1) * CDIM + c];
  return (1.f - fy) * ((1.f - fx) * v00 + fx * v01) +
         fy * ((1.f - fx) * v10 + fx * v11);
}

__global__ void epilogue_kernel(const float* __restrict__ src,
                                float* __restrict__ dst,
                                int hdim, int wdim, int up,
                                int hoff, int woff) {
  int b = blockIdx.x, wp = blockIdx.y, tid = threadIdx.x;
  int Hout = hdim << up;
  int Nn = hdim * wdim;
  const float* sb = src + (size_t)b * Nn * CDIM;
  int x0, x1; float fx;
  if (up) {
    float sx = wp * (float)(wdim - 1) / (float)(2 * wdim - 1);
    x0 = (int)floorf(sx);
    x1 = min(x0 + 1, wdim - 1);
    fx = sx - (float)x0;
  } else { x0 = x1 = wp; fx = 0.f; }
  int total = CDIM * Hout;
  float s = 0.f, ss = 0.f;
  for (int i = tid; i < total; i += 256) {
    int c = i % CDIM, hp = i / CDIM;
    float v = ep_sample(sb, hdim, wdim, up, hp, x0, x1, fx, c);
    s += v; ss += v * v;
  }
  __shared__ float rs[256], rss[256];
  rs[tid] = s; rss[tid] = ss; __syncthreads();
  for (int o = 128; o > 0; o >>= 1) {
    if (tid < o) { rs[tid] += rs[tid + o]; rss[tid] += rss[tid + o]; }
    __syncthreads();
  }
  float mu = rs[0] / (float)total;
  float var = rss[0] / (float)total - mu * mu;
  float sd = sqrtf(fmaxf(var, 0.f));
  if (sd == 0.f) sd = 1.f;
  float isd = 1.f / sd;
  for (int i = tid; i < total; i += 256) {
    int c = i % CDIM, hp = i / CDIM;
    float v = ep_sample(sb, hdim, wdim, up, hp, x0, x1, fx, c);
    dst[(((size_t)b * 64 + hoff + hp) * 64 + (woff + wp)) * CDIM + c] =
        (v - mu) * isd;
  }
}

// ---------------------------------------------------------------------------
// Host launcher
// ---------------------------------------------------------------------------
extern "C" void kernel_launch(void* const* d_in, const int* in_sizes, int n_in,
                              void* d_out, int out_size, void* d_ws, size_t ws_size,
                              hipStream_t stream) {
  (void)in_sizes; (void)n_in; (void)out_size; (void)ws_size;
  const float* x        = (const float*)d_in[0];
  // d_in[1] = Cam_Centre (unused by reference output path)
  const float* gw_Wqkv  = (const float*)d_in[2];
  const float* gw_bqkv  = (const float*)d_in[3];
  const float* as_Wqkv  = (const float*)d_in[4];
  const float* as_bqkv  = (const float*)d_in[5];
  const float* as_Wp    = (const float*)d_in[6];
  const float* as_bp    = (const float*)d_in[7];

  float* attn_all = (float*)d_out;                        // (2,64,64,384)
  float* S        = attn_all + (size_t)2 * 64 * 64 * 384; // (2,12,1024,1024)

  // workspace carve-up (256B aligned), total ~26 MB
  char* wsb = (char*)d_ws;
  size_t off = 0;
  auto take = [&](size_t bytes) -> void* {
    void* p = wsb + off;
    off = (off + bytes + 255) & ~(size_t)255;
    return p;
  };
  _Float16* xp16  = (_Float16*)take((size_t)2048 * 384 * 2);
  _Float16* qkvg  = (_Float16*)take((size_t)2048 * 1152 * 2);
  _Float16* xd16  = (_Float16*)take((size_t)2048 * 384 * 2);
  _Float16* qkvD  = (_Float16*)take((size_t)2048 * 1152 * 2);
  float*    stD   = (float*)take((size_t)24576 * 2 * 4);
  _Float16* attnD = (_Float16*)take((size_t)2048 * 384 * 2);
  float*    projD = (float*)take((size_t)2048 * 384 * 4);
  _Float16* xcrop[4]; _Float16* qkvc[4]; float* stC[4];
  _Float16* attnc[4]; float* projc[4];
  for (int i = 0; i < 4; ++i) {
    xcrop[i] = (_Float16*)take((size_t)384 * 384 * 2);
    qkvc[i]  = (_Float16*)take((size_t)384 * 1152 * 2);
    stC[i]   = (float*)take((size_t)4608 * 2 * 4);
    attnc[i] = (_Float16*)take((size_t)384 * 384 * 2);
    projc[i] = (float*)take((size_t)384 * 384 * 4);
  }

  // ---- gw branch: attn_weight ----
  pool_kernel<<<3072, 256, 0, stream>>>(x, xp16);
  gemm_bias_kernel<true><<<dim3(128, 72), 32, 0, stream>>>(
      xp16, gw_Wqkv, gw_bqkv, (void*)qkvg, 2048, 384, 1152);
  qk_scores_kernel<<<dim3(24, 64, 64), 32, 0, stream>>>(qkvg, S, 1024);
  softmax_rows_kernel<<<24576, 256, 0, stream>>>(S, 1024);

  // ---- atten_shift inputs ----
  gather_d_kernel<<<3072, 256, 0, stream>>>(x, xd16);
  const int ch0[4] = {0, 24, 8, 0};   // B1, B2, C1, C2
  const int cw0[4] = {0, 8, 0, 24};
  const int cwd[4] = {24, 24, 8, 8};
  for (int i = 0; i < 4; ++i)
    gather_crop_kernel<<<576, 256, 0, stream>>>(xp16, xcrop[i], ch0[i], cw0[i], cwd[i]);

  // ---- qkv GEMMs ----
  gemm_bias_kernel<true><<<dim3(128, 72), 32, 0, stream>>>(
      xd16, as_Wqkv, as_bqkv, (void*)qkvD, 2048, 384, 1152);
  for (int i = 0; i < 4; ++i)
    gemm_bias_kernel<true><<<dim3(24, 72), 32, 0, stream>>>(
        xcrop[i], as_Wqkv, as_bqkv, (void*)qkvc[i], 384, 384, 1152);

  // ---- streaming attention ----
  attn_pass1_kernel<<<dim3(24, 64), 32, 0, stream>>>(qkvD, stD, 1024);
  attn_pass2_kernel<<<dim3(24, 64), 32, 0, stream>>>(qkvD, stD, attnD, 1024);
  for (int i = 0; i < 4; ++i) {
    attn_pass1_kernel<<<dim3(24, 12), 32, 0, stream>>>(qkvc[i], stC[i], 192);
    attn_pass2_kernel<<<dim3(24, 12), 32, 0, stream>>>(qkvc[i], stC[i], attnc[i], 192);
  }

  // ---- projection GEMMs ----
  gemm_bias_kernel<false><<<dim3(128, 24), 32, 0, stream>>>(
      attnD, as_Wp, as_bp, (void*)projD, 2048, 384, 384);
  for (int i = 0; i < 4; ++i)
    gemm_bias_kernel<false><<<dim3(24, 24), 32, 0, stream>>>(
        attnc[i], as_Wp, as_bp, (void*)projc[i], 384, 384, 384);

  // ---- epilogue: upsample + zscore + scatter (regions partition 64x64) ----
  // B1 -> rows 0:16, cols 0:48 ; B2 -> rows 48:64, cols 16:64
  // C1 -> rows 16:64, cols 0:16 ; C2 -> rows 0:48, cols 48:64
  // D  -> rows 16:48, cols 16:48 (no upsample)
  const int ehd[4] = {8, 8, 24, 24};
  const int ewd[4] = {24, 24, 8, 8};
  const int eho[4] = {0, 48, 16, 0};
  const int ewo[4] = {0, 16, 0, 48};
  for (int i = 0; i < 4; ++i)
    epilogue_kernel<<<dim3(2, ewd[i] * 2), 256, 0, stream>>>(
        projc[i], attn_all, ehd[i], ewd[i], 1, eho[i], ewo[i]);
  epilogue_kernel<<<dim3(2, 32), 256, 0, stream>>>(
      projD, attn_all, 32, 32, 0, 16, 16);
}
